// SGDATSeg_49770081026149
// MI455X (gfx1250) — compile-verified, matrix operand order checked by reading, not run on hardware
//
#include <hip/hip_runtime.h>
#include <hip/hip_bf16.h>
#include <math.h>

typedef __attribute__((ext_vector_type(16))) _Float16 v16h;
typedef __attribute__((ext_vector_type(8)))  _Float16 v8h;
typedef __attribute__((ext_vector_type(4)))  _Float16 v4h;
typedef __attribute__((ext_vector_type(8)))  float    v8f;

#define NEGV (-1.0e9f)

#define WMMA_F16(a, b, c) \
  __builtin_amdgcn_wmma_f32_16x16x32_f16(false, (a), false, (b), (short)0, (c), false, false)

#define CAT16(lo, hi) \
  __builtin_shufflevector((lo), (hi), 0, 1, 2, 3, 4, 5, 6, 7, 8, 9, 10, 11, 12, 13, 14, 15)

// ---------------------------------------------------------------------------
// Fragment helpers for v_wmma_f32_16x16x32_f16 (wave32 layouts per ISA 7.12.2)
// A: 16x32 f16 row-major in LDS (stride ld): two contiguous 8-half runs.
// B: staged pre-swizzled fragment-major: [ks][nt][lane][16] halfs (contiguous).
// ---------------------------------------------------------------------------
__device__ inline v16h load_a_frag(const _Float16* __restrict__ sA, int ld, int kstep) {
  int lane = threadIdx.x & 31;
  int m  = lane & 15;
  int kb = ((lane >> 4) << 3) + (kstep << 5);
  const _Float16* p = sA + m * ld + kb;
  v8h lo = *(const v8h*)p;
  v8h hi = *(const v8h*)(p + 16);
  return CAT16(lo, hi);
}

__device__ inline v16h load_b_frag_swz(const _Float16* __restrict__ sW, int nn, int ks, int nt) {
  int lane = threadIdx.x & 31;
  const _Float16* p = sW + (((ks * nn + nt) * 32 + lane) << 4);
  v8h lo = *(const v8h*)p;
  v8h hi = *(const v8h*)(p + 8);
  return CAT16(lo, hi);
}

__device__ inline v8f v8f_zero() {
  v8f z;
#pragma unroll
  for (int i = 0; i < 8; ++i) z[i] = 0.0f;
  return z;
}

// ---------------------------------------------------------------------------
// Generic wave-tiled GEMM (compile-time tile shape):
//   out = [act](A*ascale @ W + bias) [+ res*res_scale]
// A: f32 (rows x ka, stride lda). W: f16 fragment-swizzled (KAPAD x NCPAD),
// optional per-batch stride. 256 threads = 8 waves, one 16-row tile per wave.
// rows % 128 == 0 for all launches.
// ---------------------------------------------------------------------------
template <int KAPAD, int NCPAD>
__global__ __launch_bounds__(256) void k_gemm(
    const float* __restrict__ A, int lda, int rows, int rows_per_batch,
    const _Float16* __restrict__ W, int w_bstride, int ka,
    const float* __restrict__ bias, int relu_flag,
    const float* __restrict__ res, int res_ld, const float* __restrict__ res_scale,
    const float* __restrict__ a_scale,
    float* __restrict__ out, int out_ld, int nc_store)
{
  constexpr int NK = KAPAD / 32, NN = NCPAD / 16;
  __shared__ _Float16 sW[KAPAD * NCPAD];
  __shared__ _Float16 sA[8][16 * KAPAD];
  const int tid = threadIdx.x, wv = tid >> 5, lane = tid & 31;
  const int blk_row0 = blockIdx.x * 128;
  const int b = blk_row0 / rows_per_batch;

  // stage W (already f16, fragment-swizzled, zero-padded): straight b128 copy
  {
    const v8h* src = (const v8h*)(W + (size_t)b * w_bstride);
    v8h* dst = (v8h*)sW;
    for (int i = tid; i < (KAPAD * NCPAD) / 8; i += 256) dst[i] = src[i];
  }

  // stage this wave's 16 x KAPAD A tile as f16 (row-major)
  const int tile0 = blk_row0 + wv * 16;
  _Float16* myA = sA[wv];
  if (ka == KAPAD) {
    // fast path: full-width rows, float4 loads -> v4h stores
    for (int idx = lane; idx < (16 * KAPAD) / 4; idx += 32) {
      int r = idx / (KAPAD / 4);
      int c4 = (idx - r * (KAPAD / 4)) * 4;
      const float4 v = *(const float4*)(A + (size_t)(tile0 + r) * lda + c4);
      float sx = 1.f, sy = 1.f, sz = 1.f, sw = 1.f;
      if (a_scale) {
        const float* as = a_scale + b * KAPAD + c4;
        sx = as[0]; sy = as[1]; sz = as[2]; sw = as[3];
      }
      v4h h;
      h[0] = (_Float16)(v.x * sx); h[1] = (_Float16)(v.y * sy);
      h[2] = (_Float16)(v.z * sz); h[3] = (_Float16)(v.w * sw);
      *(v4h*)(myA + r * KAPAD + c4) = h;
    }
  } else {
    for (int idx = lane; idx < 16 * KAPAD; idx += 32) {
      int r = idx / KAPAD, c = idx - r * KAPAD;
      float v = 0.0f;
      if (c < ka) {
        v = A[(size_t)(tile0 + r) * lda + c];
        if (a_scale) v *= a_scale[b * ka + c];
      }
      myA[idx] = (_Float16)v;
    }
  }
  __syncthreads();

  v8f acc[NN];
#pragma unroll
  for (int t = 0; t < NN; ++t) acc[t] = v8f_zero();

#pragma unroll
  for (int ks = 0; ks < NK; ++ks) {
    v16h a = load_a_frag(myA, KAPAD, ks);
#pragma unroll
    for (int t = 0; t < NN; ++t) {
      v16h bf = load_b_frag_swz(sW, NN, ks, t);
      acc[t] = WMMA_F16(a, bf, acc[t]);
    }
  }

  const int half_hi = (lane >> 4) << 3;
#pragma unroll
  for (int t = 0; t < NN; ++t) {
    int n = (t << 4) + (lane & 15);
    if (n >= nc_store) continue;
    float bv = bias ? bias[n] : 0.0f;
#pragma unroll
    for (int i = 0; i < 8; ++i) {
      int grow = tile0 + i + half_hi;
      float v = acc[t][i] + bv;
      if (relu_flag) v = fmaxf(v, 0.0f);
      if (res) {
        float rs = res_scale ? res_scale[b * 64 + n] : 1.0f;
        v += res[(size_t)grow * res_ld + n] * rs;
      }
      out[(size_t)grow * out_ld + n] = v;
    }
  }
}

// ---------------------------------------------------------------------------
// f32 -> f16 with zero padding AND fragment swizzle:
// dst linear index = ((b*NKblocks + (ks*nn+nt))*32 + lane)*16 + i
//   n = nt*16 + (lane&15);  k = ks*32 + (lane>>4)*8 + (i&7) + (i>>3)*16
// ---------------------------------------------------------------------------
__global__ void k_cvt_swz(const float* __restrict__ src, _Float16* __restrict__ dst,
                          int ka, int nc, int ka_pad, int nc_pad, int batches)
{
  const int nn = nc_pad >> 4;
  const int per = ka_pad * nc_pad;
  long long gid = (long long)blockIdx.x * 256 + threadIdx.x;
  if (gid >= (long long)per * batches) return;
  int bb = (int)(gid / per);
  int idx = (int)(gid - (long long)bb * per);
  int i = idx & 15;
  int lane = (idx >> 4) & 31;
  int rest = idx >> 9;
  int nt = rest % nn;
  int ks = rest / nn;
  int n = (nt << 4) + (lane & 15);
  int k = (ks << 5) + ((lane >> 4) << 3) + (i & 7) + ((i >> 3) << 4);
  float v = (k < ka && n < nc) ? src[(size_t)bb * ka * nc + (size_t)k * nc + n] : 0.0f;
  dst[gid] = (_Float16)v;
}

// ---------------------------------------------------------------------------
// Farthest point sampling: one 1024-thread workgroup per batch, coords +
// running min-distance fully in LDS (128 KB of the 320 KB WGP pool).
// Level-2 (2048) indices are a prefix of the 4096-long sequence.
// ---------------------------------------------------------------------------
#define NPTS 8192
__global__ __launch_bounds__(1024) void k_fps(const float* __restrict__ x,
                                              int* __restrict__ fps_idx, int M)
{
  __shared__ float px[NPTS], py[NPTS], pz[NPTS], dist[NPTS];
  __shared__ float rv[32];
  __shared__ int   ri[32];
  __shared__ int   bcast;
  const int b = blockIdx.x, tid = threadIdx.x;

  for (int i = tid; i < NPTS; i += 1024) {
    const float* p = x + ((size_t)b * NPTS + i) * 9;
    px[i] = p[0]; py[i] = p[1]; pz[i] = p[2];
    dist[i] = 1.0e10f;
  }
  if (tid == 0) { fps_idx[b * M] = 0; bcast = 0; }
  __syncthreads();

  int last = 0;
  for (int s = 1; s < M; ++s) {
    float lx = px[last], ly = py[last], lz = pz[last];
    float best = -1.0f; int bi = 0x7fffffff;
    for (int i = tid; i < NPTS; i += 1024) {
      float dx = px[i] - lx, dy = py[i] - ly, dz = pz[i] - lz;
      float nd = fminf(dist[i], dx * dx + dy * dy + dz * dz);
      dist[i] = nd;
      if (nd > best) { best = nd; bi = i; }
    }
    for (int off = 16; off > 0; off >>= 1) {
      float ov = __shfl_xor(best, off);
      int   oi = __shfl_xor(bi, off);
      if (ov > best || (ov == best && oi < bi)) { best = ov; bi = oi; }
    }
    if ((tid & 31) == 0) { rv[tid >> 5] = best; ri[tid >> 5] = bi; }
    __syncthreads();
    if (tid < 32) {
      best = rv[tid]; bi = ri[tid];
      for (int off = 16; off > 0; off >>= 1) {
        float ov = __shfl_xor(best, off);
        int   oi = __shfl_xor(bi, off);
        if (ov > best || (ov == best && oi < bi)) { best = ov; bi = oi; }
      }
      if (tid == 0) { bcast = bi; fps_idx[b * M + s] = bi; }
    }
    __syncthreads();
    last = bcast;
  }
}

// ---------------------------------------------------------------------------
// KNN (K=32) + radius mask: one wave per centroid. Per-lane top-32 kept in LDS
// via guarded max-replace over a 256-point strided scan, then 32 rounds of
// packed u64 shuffle min-reduce to extract the global 32 nearest.
// ---------------------------------------------------------------------------
__global__ __launch_bounds__(256) void k_knn(
    const float* __restrict__ x, const int* __restrict__ fps_idx,
    int m, float rmin, float rmax,
    int* __restrict__ knn, unsigned* __restrict__ maskw)
{
  __shared__ float cand_d[256 * 32];
  __shared__ int   cand_i[256 * 32];
  const int b = blockIdx.y, tid = threadIdx.x, wv = tid >> 5, lane = tid & 31;
  const int ci = blockIdx.x * 8 + wv;
  const int cpt = fps_idx[b * 4096 + ci];
  const float* cp = x + ((size_t)b * NPTS + cpt) * 9;
  const float cx = cp[0], cy = cp[1], cz = cp[2];

  float* myd = cand_d + tid * 32;
  int*   myi = cand_i + tid * 32;

  int cnt = 0, maxslot = 0;
  float curmax = -1.0f;
  for (int p = lane; p < NPTS; p += 32) {
    const float* pp = x + ((size_t)b * NPTS + p) * 9;
    float dx = pp[0] - cx, dy = pp[1] - cy, dz = pp[2] - cz;
    float d2 = dx * dx + dy * dy + dz * dz;
    if (cnt < 32) {
      myd[cnt] = d2; myi[cnt] = p;
      if (d2 > curmax) { curmax = d2; maxslot = cnt; }
      cnt++;
    } else if (d2 < curmax) {
      myd[maxslot] = d2; myi[maxslot] = p;
      curmax = -1.0f;
      for (int s = 0; s < 32; ++s) {
        float v = myd[s];
        if (v > curmax) { curmax = v; maxslot = s; }
      }
    }
  }

  const int gbase = b * m + ci;
  float mydist = 0.0f;
  for (int r = 0; r < 32; ++r) {
    float lmin = 3.4e38f; int lslot = 0;
    for (int s = 0; s < 32; ++s) {
      float v = myd[s];
      if (v < lmin) { lmin = v; lslot = s; }
    }
    unsigned li = (unsigned)myi[lslot] & 0xFFFFu;
    unsigned long long key =
        ((unsigned long long)__float_as_uint(lmin) << 32) |
        li | ((unsigned)lslot << 16) | ((unsigned)lane << 21);
    for (int off = 16; off > 0; off >>= 1) {
      unsigned long long ok = __shfl_xor(key, off);
      if (ok < key) key = ok;
    }
    float    wd2   = __uint_as_float((unsigned)(key >> 32));
    unsigned low   = (unsigned)key;
    int      widx  = low & 0xFFFF;
    int      wslot = (low >> 16) & 31;
    int      wlane = (low >> 21) & 31;
    if (lane == wlane) myd[wslot] = 3.4e38f;
    if (lane == r) {
      mydist = sqrtf(fmaxf(wd2, 1e-12f));
      knn[(size_t)gbase * 32 + r] = widx;
    }
  }
  float s = mydist;
  for (int off = 16; off > 0; off >>= 1) s += __shfl_xor(s, off);
  float rr = fminf(fmaxf(s * (1.0f / 32.0f), rmin), rmax);
  unsigned long long bal = __ballot(mydist <= rr);
  if (lane == 0) maskw[gbase] = (unsigned)bal;
}

// ---------------------------------------------------------------------------
// Fuse: per centroid (one wave): gather 32 neighbor feats, (neigh-cf)->f16,
// 16x WMMA with wn (fragment-swizzled), bias+ReLU, masked max-pool via
// shfl_xor(16), sigmoid gate, residual add.
// ---------------------------------------------------------------------------
__global__ __launch_bounds__(256) void k_fuse(
    const float* __restrict__ feats, const int* __restrict__ fps_idx,
    const int* __restrict__ knn, const unsigned* __restrict__ maskw,
    const _Float16* __restrict__ wn16, const float* __restrict__ bn,
    const float* __restrict__ wg, const float* __restrict__ bg,
    int m, float* __restrict__ outl)
{
  __shared__ _Float16 sWn[64 * 64];
  __shared__ float    sWg[64 * 64];
  __shared__ float    sBn[64], sBg[64];
  __shared__ _Float16 sA[8][32 * 64];
  __shared__ float    sCf[8][64];
  __shared__ float    sPool[8][64];
  const int tid = threadIdx.x, wv = tid >> 5, lane = tid & 31, b = blockIdx.y;

  {
    const v8h* src = (const v8h*)wn16;
    v8h* dst = (v8h*)sWn;
    for (int i = tid; i < (64 * 64) / 8; i += 256) dst[i] = src[i];
  }
  for (int i = tid; i < 64 * 64; i += 256) sWg[i] = wg[i];
  if (tid < 64) { sBn[tid] = bn[tid]; sBg[tid] = bg[tid]; }

  const int ci = blockIdx.x * 8 + wv;
  const int gbase = b * m + ci;
  const int cpt = fps_idx[b * 4096 + ci];
  const float* cf = feats + ((size_t)b * NPTS + cpt) * 64;
  sCf[wv][lane]      = cf[lane];
  sCf[wv][lane + 32] = cf[lane + 32];
  __syncthreads();

  // stage A rows (32 x 64 f16): lane owns neighbor row `lane`
  const int nb = knn[(size_t)gbase * 32 + lane];
  const float4* nfv = (const float4*)(feats + ((size_t)b * NPTS + nb) * 64);
  const float4* cfv = (const float4*)(&sCf[wv][0]);
#pragma unroll
  for (int c4 = 0; c4 < 16; ++c4) {
    float4 a = nfv[c4];
    float4 c = cfv[c4];
    v4h h;
    h[0] = (_Float16)(a.x - c.x); h[1] = (_Float16)(a.y - c.y);
    h[2] = (_Float16)(a.z - c.z); h[3] = (_Float16)(a.w - c.w);
    *(v4h*)(sA[wv] + lane * 64 + c4 * 4) = h;
  }
  __syncthreads();

  const unsigned mask = maskw[gbase];

  v16h bf[2][4];
#pragma unroll
  for (int ks = 0; ks < 2; ++ks)
#pragma unroll
    for (int t = 0; t < 4; ++t) bf[ks][t] = load_b_frag_swz(sWn, 4, ks, t);

  float pooled[4] = {NEGV, NEGV, NEGV, NEGV};
#pragma unroll
  for (int mt = 0; mt < 2; ++mt) {
    const _Float16* Ap = sA[wv] + mt * 16 * 64;
    v16h a0 = load_a_frag(Ap, 64, 0);
    v16h a1 = load_a_frag(Ap, 64, 1);
#pragma unroll
    for (int t = 0; t < 4; ++t) {
      v8f acc = v8f_zero();
      acc = WMMA_F16(a0, bf[0][t], acc);
      acc = WMMA_F16(a1, bf[1][t], acc);
      int n = (t << 4) + (lane & 15);
      float bv = sBn[n];
      float mx = NEGV;
#pragma unroll
      for (int i = 0; i < 8; ++i) {
        int rowk = mt * 16 + i + ((lane >> 4) << 3);
        float h = fmaxf(acc[i] + bv, 0.0f);
        float hm = ((mask >> rowk) & 1u) ? h : NEGV;
        mx = fmaxf(mx, hm);
      }
      mx = fmaxf(mx, __shfl_xor(mx, 16));
      pooled[t] = fmaxf(pooled[t], mx);
    }
  }
  if (lane < 16) {
#pragma unroll
    for (int t = 0; t < 4; ++t) sPool[wv][(t << 4) + lane] = pooled[t];
  }
  __syncthreads();

  for (int cc = lane; cc < 64; cc += 32) {
    float a = sBg[cc];
    for (int k = 0; k < 64; ++k) a += sCf[wv][k] * sWg[k * 64 + cc];
    float g = 1.0f / (1.0f + expf(-a));
    outl[(size_t)gbase * 64 + cc] = g * sPool[wv][cc] + sCf[wv][cc];
  }
}

// ---------------------------------------------------------------------------
// CCC channel attention: a = sigmoid(relu(mean(x,1)@w1+b1)@w2+b2)  (B x 64)
// ---------------------------------------------------------------------------
__global__ __launch_bounds__(256) void k_ccc(
    const float* __restrict__ xin, int m,
    const float* __restrict__ w1, const float* __restrict__ b1,
    const float* __restrict__ w2, const float* __restrict__ b2,
    float* __restrict__ aout)
{
  __shared__ float part[256];
  __shared__ float smean[64];
  __shared__ float shid[32];
  const int b = blockIdx.x, tid = threadIdx.x;
  const int c = tid & 63, q = tid >> 6;
  float s = 0.0f;
  for (int r = q; r < m; r += 4) s += xin[((size_t)b * m + r) * 64 + c];
  part[tid] = s;
  __syncthreads();
  if (tid < 64) smean[tid] = (part[tid] + part[tid + 64] + part[tid + 128] + part[tid + 192]) / (float)m;
  __syncthreads();
  if (tid < 32) {
    float a = b1[tid];
    for (int k = 0; k < 64; ++k) a += smean[k] * w1[k * 32 + tid];
    shid[tid] = fmaxf(a, 0.0f);
  }
  __syncthreads();
  if (tid < 64) {
    float a = b2[tid];
    for (int k = 0; k < 32; ++k) a += shid[k] * w2[k * 64 + tid];
    aout[b * 64 + tid] = 1.0f / (1.0f + expf(-a));
  }
}

// softmax over 64 channels per row, in place; one wave per row
__global__ __launch_bounds__(256) void k_rowsoftmax(float* __restrict__ q, int rows)
{
  const int tid = threadIdx.x, wv = tid >> 5, lane = tid & 31;
  const int row = blockIdx.x * 8 + wv;
  if (row >= rows) return;
  float* p = q + (size_t)row * 64;
  float v0 = p[lane], v1 = p[lane + 32];
  float mx = fmaxf(v0, v1);
  for (int off = 16; off > 0; off >>= 1) mx = fmaxf(mx, __shfl_xor(mx, off));
  float e0 = expf(v0 - mx), e1 = expf(v1 - mx);
  float s = e0 + e1;
  for (int off = 16; off > 0; off >>= 1) s += __shfl_xor(s, off);
  float inv = 1.0f / s;
  p[lane] = e0 * inv; p[lane + 32] = e1 * inv;
}

// per-(batch, col) max and sum(exp) over m rows for the point-axis softmax
__global__ __launch_bounds__(128) void k_colstats(
    const float* __restrict__ kr, int m,
    float* __restrict__ cmax, float* __restrict__ csum)
{
  __shared__ float red[128];
  const int c = blockIdx.x, b = blockIdx.y, tid = threadIdx.x;
  const float* base = kr + (size_t)b * m * 64 + c;
  float mx = -3.4e38f;
  for (int r = tid; r < m; r += 128) mx = fmaxf(mx, base[(size_t)r * 64]);
  red[tid] = mx; __syncthreads();
  for (int off = 64; off > 0; off >>= 1) { if (tid < off) red[tid] = fmaxf(red[tid], red[tid + off]); __syncthreads(); }
  const float M = red[0];
  __syncthreads();
  float s = 0.0f;
  for (int r = tid; r < m; r += 128) s += expf(base[(size_t)r * 64] - M);
  red[tid] = s; __syncthreads();
  for (int off = 64; off > 0; off >>= 1) { if (tid < off) red[tid] += red[tid + off]; __syncthreads(); }
  if (tid == 0) { cmax[b * 64 + c] = M; csum[b * 64 + c] = red[0]; }
}

// ctx[b,c,d] = sum_m softmax_col(k)[m,c] * v[m,d]; block per (c,b), 64 threads
__global__ __launch_bounds__(64) void k_ctx(
    const float* __restrict__ kr, const float* __restrict__ v, int m,
    const float* __restrict__ cmax, const float* __restrict__ csum,
    float* __restrict__ ctx)
{
  __shared__ float kn[64];
  const int c = blockIdx.x, b = blockIdx.y, t = threadIdx.x;
  const float M = cmax[b * 64 + c], inv = 1.0f / csum[b * 64 + c];
  float acc = 0.0f;
  for (int m0 = 0; m0 < m; m0 += 64) {
    kn[t] = expf(kr[((size_t)b * m + m0 + t) * 64 + c] - M) * inv;
    __syncthreads();
    for (int j = 0; j < 64; ++j) acc += kn[j] * v[((size_t)b * m + m0 + j) * 64 + t];
    __syncthreads();
  }
  ctx[((size_t)b * 64 + c) * 64 + t] = acc;
}

// fused[b, i, :] = concat(y1[b, i>>1, :], y2[b, i>>2, :])
__global__ void k_concat(const float* __restrict__ y1, const float* __restrict__ y2,
                         float* __restrict__ fused)
{
  size_t i = (size_t)blockIdx.x * 256 + threadIdx.x;
  const size_t total = (size_t)4 * 8192 * 128;
  if (i >= total) return;
  int c = (int)(i & 127);
  size_t pi = i >> 7;
  int pt = (int)(pi & 8191);
  int b = (int)(pi >> 13);
  float v;
  if (c < 64) v = y1[((size_t)b * 4096 + (pt >> 1)) * 64 + c];
  else        v = y2[((size_t)b * 2048 + (pt >> 2)) * 64 + (c - 64)];
  fused[i] = v;
}

// ---------------------------------------------------------------------------
extern "C" void kernel_launch(void* const* d_in, const int* in_sizes, int n_in,
                              void* d_out, int out_size, void* d_ws, size_t ws_size,
                              hipStream_t stream) {
  const int B = 4, N = 8192, M1 = 4096, M2 = 2048;
  const float* x      = (const float*)d_in[0];
  const float* enc_w1 = (const float*)d_in[1];
  const float* enc_b1 = (const float*)d_in[2];
  const float* enc_w2 = (const float*)d_in[3];
  const float* enc_b2 = (const float*)d_in[4];
  const float* f1_wn  = (const float*)d_in[5];
  const float* f1_bn  = (const float*)d_in[6];
  const float* f1_wg  = (const float*)d_in[7];
  const float* f1_bg  = (const float*)d_in[8];
  const float* f2_wn  = (const float*)d_in[9];
  const float* f2_bn  = (const float*)d_in[10];
  const float* f2_wg  = (const float*)d_in[11];
  const float* f2_bg  = (const float*)d_in[12];
  const float* ccc_w1 = (const float*)d_in[13];
  const float* ccc_b1 = (const float*)d_in[14];
  const float* ccc_w2 = (const float*)d_in[15];
  const float* ccc_b2 = (const float*)d_in[16];
  const float* gva_wq = (const float*)d_in[17];
  const float* gva_wk = (const float*)d_in[18];
  const float* gva_wv = (const float*)d_in[19];
  const float* gva_wo = (const float*)d_in[20];
  const float* up_w1  = (const float*)d_in[21];
  const float* up_b1  = (const float*)d_in[22];
  const float* up_w2  = (const float*)d_in[23];
  const float* up_b2  = (const float*)d_in[24];
  const float* cls_w1 = (const float*)d_in[25];
  const float* cls_b1 = (const float*)d_in[26];
  const float* cls_w2 = (const float*)d_in[27];
  const float* cls_b2 = (const float*)d_in[28];
  float* dout = (float*)d_out;

  // ---- workspace carve-out ----
  size_t off = 0;
  auto alloc = [&](size_t bytes) -> char* {
    char* p = (char*)d_ws + off;
    off += (bytes + 255) & ~(size_t)255;
    return p;
  };
  _Float16* wf_enc1 = (_Float16*)alloc(32 * 64 * 2);
  _Float16* wf_enc2 = (_Float16*)alloc(64 * 64 * 2);
  _Float16* wf_f1wn = (_Float16*)alloc(64 * 64 * 2);
  _Float16* wf_f2wn = (_Float16*)alloc(64 * 64 * 2);
  _Float16* wf_q    = (_Float16*)alloc(64 * 64 * 2);
  _Float16* wf_k    = (_Float16*)alloc(64 * 64 * 2);
  _Float16* wf_v    = (_Float16*)alloc(64 * 64 * 2);
  _Float16* wf_o    = (_Float16*)alloc(64 * 64 * 2);
  _Float16* wf_up1  = (_Float16*)alloc(128 * 64 * 2);
  _Float16* wf_up2  = (_Float16*)alloc(64 * 64 * 2);
  _Float16* wf_cls1 = (_Float16*)alloc(64 * 32 * 2);
  _Float16* wf_cls2 = (_Float16*)alloc(32 * 16 * 2);
  _Float16* ctx16   = (_Float16*)alloc((size_t)B * 64 * 64 * 2);

  float* feats  = (float*)alloc((size_t)B * N * 64 * 4);     // later reused as up-hidden
  float* hidenc = (float*)alloc((size_t)B * N * 64 * 4);     // later reused as up2 out
  int*   fpsi   = (int*)alloc((size_t)B * M1 * 4);
  int*   knn1   = (int*)alloc((size_t)B * M1 * 32 * 4);
  unsigned* msk1 = (unsigned*)alloc((size_t)B * M1 * 4);
  int*   knn2   = (int*)alloc((size_t)B * M2 * 32 * 4);
  unsigned* msk2 = (unsigned*)alloc((size_t)B * M2 * 4);
  float* out1   = (float*)alloc((size_t)B * M1 * 64 * 4);
  float* out2   = (float*)alloc((size_t)B * M2 * 64 * 4);
  float* abuf   = (float*)alloc((size_t)B * 64 * 4);
  float* qbuf   = (float*)alloc((size_t)B * M1 * 64 * 4);    // later reused as cls hidden
  float* kbuf   = (float*)alloc((size_t)B * M1 * 64 * 4);
  float* vbuf   = (float*)alloc((size_t)B * M1 * 64 * 4);
  float* t1buf  = (float*)alloc((size_t)B * M1 * 64 * 4);
  float* cmax   = (float*)alloc((size_t)B * 64 * 4);
  float* csum   = (float*)alloc((size_t)B * 64 * 4);
  float* ctxf   = (float*)alloc((size_t)B * 64 * 64 * 4);
  float* y1     = (float*)alloc((size_t)B * M1 * 64 * 4);
  float* y2     = (float*)alloc((size_t)B * M2 * 64 * 4);
  float* fused  = (float*)alloc((size_t)B * N * 128 * 4);
  (void)ws_size; (void)in_sizes; (void)n_in; (void)out_size;

  auto cvt = [&](const float* s, _Float16* d, int ka, int nc, int kp, int np, int batches) {
    long long tot = (long long)kp * np * batches;
    k_cvt_swz<<<(int)((tot + 255) / 256), 256, 0, stream>>>(s, d, ka, nc, kp, np, batches);
  };
  cvt(enc_w1, wf_enc1, 9, 64, 32, 64, 1);
  cvt(enc_w2, wf_enc2, 64, 64, 64, 64, 1);
  cvt(f1_wn, wf_f1wn, 64, 64, 64, 64, 1);
  cvt(f2_wn, wf_f2wn, 64, 64, 64, 64, 1);
  cvt(gva_wq, wf_q, 64, 64, 64, 64, 1);
  cvt(gva_wk, wf_k, 64, 64, 64, 64, 1);
  cvt(gva_wv, wf_v, 64, 64, 64, 64, 1);
  cvt(gva_wo, wf_o, 64, 64, 64, 64, 1);
  cvt(up_w1, wf_up1, 128, 64, 128, 64, 1);
  cvt(up_w2, wf_up2, 64, 64, 64, 64, 1);
  cvt(cls_w1, wf_cls1, 64, 32, 64, 32, 1);
  cvt(cls_w2, wf_cls2, 32, 13, 32, 16, 1);

  auto gemm = [&](const float* A, int lda, int rows, int rpb,
                  const _Float16* W, int wbs, int ka, int kapad, int ncpad,
                  const float* bias, int relu,
                  const float* res, int resld, const float* rscale,
                  const float* ascale, float* out, int outld, int ncstore) {
    dim3 g(rows / 128);
    if (kapad == 32 && ncpad == 64)
      k_gemm<32, 64><<<g, 256, 0, stream>>>(A, lda, rows, rpb, W, wbs, ka, bias, relu, res, resld, rscale, ascale, out, outld, ncstore);
    else if (kapad == 64 && ncpad == 64)
      k_gemm<64, 64><<<g, 256, 0, stream>>>(A, lda, rows, rpb, W, wbs, ka, bias, relu, res, resld, rscale, ascale, out, outld, ncstore);
    else if (kapad == 128 && ncpad == 64)
      k_gemm<128, 64><<<g, 256, 0, stream>>>(A, lda, rows, rpb, W, wbs, ka, bias, relu, res, resld, rscale, ascale, out, outld, ncstore);
    else if (kapad == 64 && ncpad == 32)
      k_gemm<64, 32><<<g, 256, 0, stream>>>(A, lda, rows, rpb, W, wbs, ka, bias, relu, res, resld, rscale, ascale, out, outld, ncstore);
    else if (kapad == 32 && ncpad == 16)
      k_gemm<32, 16><<<g, 256, 0, stream>>>(A, lda, rows, rpb, W, wbs, ka, bias, relu, res, resld, rscale, ascale, out, outld, ncstore);
  };

  // encoder: feats = relu(x@w1+b1)@w2+b2
  gemm(x, 9, B * N, N, wf_enc1, 0, 9, 32, 64, enc_b1, 1,
       nullptr, 0, nullptr, nullptr, hidenc, 64, 64);
  gemm(hidenc, 64, B * N, N, wf_enc2, 0, 64, 64, 64, enc_b2, 0,
       nullptr, 0, nullptr, nullptr, feats, 64, 64);

  // FPS (level-2 indices are a prefix of level-1)
  k_fps<<<B, 1024, 0, stream>>>(x, fpsi, M1);

  // KNN + radius masks
  k_knn<<<dim3(M1 / 8, B), 256, 0, stream>>>(x, fpsi, M1, 0.02f, 0.15f, knn1, msk1);
  k_knn<<<dim3(M2 / 8, B), 256, 0, stream>>>(x, fpsi, M2, 0.05f, 0.30f, knn2, msk2);

  // fuse
  k_fuse<<<dim3(M1 / 8, B), 256, 0, stream>>>(feats, fpsi, knn1, msk1, wf_f1wn, f1_bn, f1_wg, f1_bg, M1, out1);
  k_fuse<<<dim3(M2 / 8, B), 256, 0, stream>>>(feats, fpsi, knn2, msk2, wf_f2wn, f2_bn, f2_wg, f2_bg, M2, out2);

  // ccc + gva per level
  auto level = [&](float* outl, int m, float* ybuf) {
    k_ccc<<<B, 256, 0, stream>>>(outl, m, ccc_w1, ccc_b1, ccc_w2, ccc_b2, abuf);
    // q = softmax_row((x*a) @ wq)
    gemm(outl, 64, B * m, m, wf_q, 0, 64, 64, 64, nullptr, 0,
         nullptr, 0, nullptr, abuf, qbuf, 64, 64);
    k_rowsoftmax<<<(B * m) / 8, 256, 0, stream>>>(qbuf, B * m);
    // k raw, col stats
    gemm(outl, 64, B * m, m, wf_k, 0, 64, 64, 64, nullptr, 0,
         nullptr, 0, nullptr, abuf, kbuf, 64, 64);
    k_colstats<<<dim3(64, B), 128, 0, stream>>>(kbuf, m, cmax, csum);
    // v
    gemm(outl, 64, B * m, m, wf_v, 0, 64, 64, 64, nullptr, 0,
         nullptr, 0, nullptr, abuf, vbuf, 64, 64);
    // ctx = softmax_col(k)^T v, then f16 (fragment-swizzled, per-batch)
    k_ctx<<<dim3(64, B), 64, 0, stream>>>(kbuf, vbuf, m, cmax, csum, ctxf);
    cvt(ctxf, ctx16, 64, 64, 64, 64, B);
    // t1 = q @ ctx[b]  (per-batch W stride)
    gemm(qbuf, 64, B * m, m, ctx16, 64 * 64, 64, 64, 64, nullptr, 0,
         nullptr, 0, nullptr, nullptr, t1buf, 64, 64);
    // y = (x*a) + t1 @ wo
    gemm(t1buf, 64, B * m, m, wf_o, 0, 64, 64, 64, nullptr, 0,
         outl, 64, abuf, nullptr, ybuf, 64, 64);
  };
  level(out1, M1, y1);
  level(out2, M2, y2);

  // upsample + concat
  k_concat<<<(int)(((size_t)B * N * 128 + 255) / 256), 256, 0, stream>>>(y1, y2, fused);

  // up MLP (reuse feats / hidenc), classifier head
  gemm(fused, 128, B * N, N, wf_up1, 0, 128, 128, 64, up_b1, 1,
       nullptr, 0, nullptr, nullptr, feats, 64, 64);
  gemm(feats, 64, B * N, N, wf_up2, 0, 64, 64, 64, up_b2, 0,
       nullptr, 0, nullptr, nullptr, hidenc, 64, 64);
  gemm(hidenc, 64, B * N, N, wf_cls1, 0, 64, 64, 32, cls_b1, 1,
       nullptr, 0, nullptr, nullptr, qbuf, 32, 32);
  gemm(qbuf, 32, B * N, N, wf_cls2, 0, 32, 32, 16, cls_b2, 0,
       nullptr, 0, nullptr, nullptr, dout, 13, 13);
}
